// CoAttention_8589934709
// MI455X (gfx1250) — matrix operand order repaired and checked by reading, not compiled, not run on hardware
//
#include <hip/hip_runtime.h>

typedef __attribute__((ext_vector_type(16))) __bf16       v16bf;
typedef __attribute__((ext_vector_type(8)))  float        v8f;
typedef __attribute__((ext_vector_type(4)))  float        f32x4;
typedef __attribute__((ext_vector_type(4)))  unsigned int u32x4;
typedef __attribute__((ext_vector_type(8)))  unsigned int u32x8;
typedef __attribute__((ext_vector_type(2)))  unsigned int u32x2;

#define LQ 32              // query rows per workgroup
#define CK 32              // keys per chunk
#define DD 1024            // head dim
#define LKEYS 2048
#define NCHUNK (LKEYS / CK)

__device__ __forceinline__ unsigned short f2bf(float f) {
  unsigned u = __builtin_bit_cast(unsigned, f);
  u += 0x7fffu + ((u >> 16) & 1u);           // round-to-nearest-even
  return (unsigned short)(u >> 16);
}

// A-fragment (16x32 bf16): lanes 0-15 -> M=0..15, K 0..15; lanes 16-31 -> K 16..31
__device__ __forceinline__ v16bf ldsA(const unsigned short* base, int stride,
                                      int r0, int k0, int lane) {
  int r  = r0 + (lane & 15);
  int kb = k0 + ((lane & 16) ? 16 : 0);
  const unsigned short* p = base + r * stride + kb;
  union { v16bf v; u32x4 q[2]; } U;
  U.q[0] = *(const u32x4*)p;
  U.q[1] = *(const u32x4*)(p + 8);
  return U.v;
}

// B-fragment (32x16 bf16) from [N][K] row-major LDS: lanes 0-15 -> N, K 0..15; lanes 16-31 -> K 16..31
__device__ __forceinline__ v16bf ldsB(const unsigned short* base, int stride,
                                      int n0, int k0, int lane) {
  int n  = n0 + (lane & 15);
  int kb = k0 + ((lane & 16) ? 16 : 0);
  const unsigned short* p = base + n * stride + kb;
  union { v16bf v; u32x8 u; } U;
  U.u = *(const u32x8*)p;
  return U.v;
}

__device__ __forceinline__ v8f wmma_bf16(v16bf a, v16bf b, v8f c) {
  return __builtin_amdgcn_wmma_f32_16x16x32_bf16(false, a, false, b, (short)0, c,
                                                 false, false);
}

__global__ __launch_bounds__(256, 1)
void coattn_flash(const float* __restrict__ S, const float* __restrict__ D,
                  float* __restrict__ out) {
  extern __shared__ char smem_raw[];
  unsigned short* Qb = (unsigned short*)smem_raw;          // [32][1024] bf16
  unsigned short* Kb = Qb + LQ * DD;                       // [32][1024] bf16 (key-major)
  unsigned short* VT = Kb + CK * DD;                       // [1024][32] bf16 (d-major)
  float* scp          = (float*)(VT + DD * CK);            // [2][32][32] split-K partials
  unsigned short* Pb  = (unsigned short*)(scp + 2 * LQ * CK); // [32][32] bf16 probs
  float* mrow = (float*)(Pb + LQ * CK);                    // running max
  float* lrow = mrow + LQ;                                 // running sum
  float* srow = lrow + LQ;                                 // per-chunk rescale

  const int tid  = threadIdx.x;
  const int lane = tid & 31;
  const int wv   = tid >> 5;
  const int b    = blockIdx.y;
  const int dir  = blockIdx.z;
  const int q0   = blockIdx.x * LQ;

  const float* Qg  = dir ? S : D;
  const float* KVg = dir ? D : S;
  float* Og = out + (size_t)dir * ((size_t)8 * LKEYS * DD);

  const float* qsrc  = Qg  + ((size_t)b * LKEYS + q0) * DD;
  const float* kvsrc = KVg + (size_t)b * LKEYS * DD;

  // ---- stage Q tile into LDS as bf16 ----
  {
    int c4 = tid * 4;
#pragma unroll
    for (int it = 0; it < LQ / 2; ++it) {
      int r = it * 2;
      f32x4 x = *(const f32x4*)(qsrc + (size_t)r * DD + c4);
      f32x4 y = *(const f32x4*)(qsrc + (size_t)(r + 1) * DD + c4);
      u32x2 px, py;
      px[0] = (unsigned)f2bf(x[0]) | ((unsigned)f2bf(x[1]) << 16);
      px[1] = (unsigned)f2bf(x[2]) | ((unsigned)f2bf(x[3]) << 16);
      py[0] = (unsigned)f2bf(y[0]) | ((unsigned)f2bf(y[1]) << 16);
      py[1] = (unsigned)f2bf(y[2]) | ((unsigned)f2bf(y[3]) << 16);
      *(u32x2*)(Qb + r * DD + c4)       = px;
      *(u32x2*)(Qb + (r + 1) * DD + c4) = py;
    }
  }
  if (tid < LQ) { mrow[tid] = -3.0e38f; lrow[tid] = 0.0f; }
  __syncthreads();

  // ---- per-wave score assignment: (row-tile, key-tile, d-half) ----
  const int st  = wv >> 1;
  const int srt = st >> 1;          // score row tile (q)
  const int sct = st & 1;           // score col tile (key)
  const int d0  = (wv & 1) * (DD / 2);
  const int hi8 = (lane & 16) ? 8 : 0;

  // hoist Q A-fragments for this wave's d-half (persistent VGPRs)
  v16bf qf[16];
#pragma unroll
  for (int ks = 0; ks < 16; ++ks)
    qf[ks] = ldsA(Qb, DD, srt * 16, d0 + ks * 32, lane);

  const int c0 = wv * 128;          // this wave's output-column slice
  v8f vzero = {0.f, 0.f, 0.f, 0.f, 0.f, 0.f, 0.f, 0.f};
  v8f acc0[8], acc1[8];
#pragma unroll
  for (int i = 0; i < 8; ++i) { acc0[i] = vzero; acc1[i] = vzero; }

  for (int kc = 0; kc < NCHUNK; ++kc) {
    // ---- load K/V chunk: row-major Kb (for QK^T) and transposed VT (for PV) ----
    {
      const float* src = kvsrc + (size_t)kc * CK * DD;
      int c4 = tid * 4;
#pragma unroll
      for (int it = 0; it < CK / 2; ++it) {
        int k = it * 2;
        f32x4 x = *(const f32x4*)(src + (size_t)k * DD + c4);
        f32x4 y = *(const f32x4*)(src + (size_t)(k + 1) * DD + c4);
        unsigned short bx0 = f2bf(x[0]), bx1 = f2bf(x[1]), bx2 = f2bf(x[2]), bx3 = f2bf(x[3]);
        unsigned short by0 = f2bf(y[0]), by1 = f2bf(y[1]), by2 = f2bf(y[2]), by3 = f2bf(y[3]);
        u32x2 px, py;
        px[0] = (unsigned)bx0 | ((unsigned)bx1 << 16);
        px[1] = (unsigned)bx2 | ((unsigned)bx3 << 16);
        py[0] = (unsigned)by0 | ((unsigned)by1 << 16);
        py[1] = (unsigned)by2 | ((unsigned)by3 << 16);
        *(u32x2*)(Kb + k * DD + c4)       = px;
        *(u32x2*)(Kb + (k + 1) * DD + c4) = py;
        *(unsigned int*)(VT + (c4 + 0) * CK + k) = (unsigned)bx0 | ((unsigned)by0 << 16);
        *(unsigned int*)(VT + (c4 + 1) * CK + k) = (unsigned)bx1 | ((unsigned)by1 << 16);
        *(unsigned int*)(VT + (c4 + 2) * CK + k) = (unsigned)bx2 | ((unsigned)by2 << 16);
        *(unsigned int*)(VT + (c4 + 3) * CK + k) = (unsigned)bx3 | ((unsigned)by3 << 16);
        if (kc + 1 < NCHUNK)
          __builtin_prefetch(src + (size_t)CK * DD + (size_t)k * DD + c4, 0, 0);
      }
    }
    __syncthreads();

    // ---- scores: each wave one 16x16 tile over its 512-wide d-half ----
    {
      v8f c = vzero;
#pragma unroll
      for (int ks = 0; ks < 16; ++ks) {
        v16bf bv = ldsB(Kb, DD, sct * 16, d0 + ks * 32, lane);
        c = wmma_bf16(qf[ks], bv, c);
      }
      float* sp = scp + (wv & 1) * (LQ * CK);
      int col = sct * 16 + (lane & 15);
      int rb  = srt * 16 + hi8;
#pragma unroll
      for (int i = 0; i < 8; ++i) sp[(rb + i) * CK + col] = c[i];
    }
    __syncthreads();

    // ---- online softmax: one thread per query row, sum the two d-half partials ----
    if (tid < LQ) {
      int r = tid;
      float s[CK];
      float cm = -3.0e38f;
#pragma unroll
      for (int k = 0; k < CK; ++k) {
        s[k] = scp[r * CK + k] + scp[LQ * CK + r * CK + k];
        cm = fmaxf(cm, s[k]);
      }
      float mo = mrow[r];
      float mn = fmaxf(mo, cm);
      float f  = __expf(mo - mn);
      float sum = 0.f;
#pragma unroll
      for (int k = 0; k < CK; k += 2) {
        float p0 = __expf(s[k]     - mn);
        float p1 = __expf(s[k + 1] - mn);
        sum += p0 + p1;
        *(unsigned int*)(Pb + r * CK + k) =
            (unsigned)f2bf(p0) | ((unsigned)f2bf(p1) << 16);
      }
      mrow[r] = mn;
      lrow[r] = lrow[r] * f + sum;
      srow[r] = f;
    }
    __syncthreads();

    // ---- rescale accumulators + PV into this wave's 128-column slice ----
    {
      float s0[8], s1[8];
#pragma unroll
      for (int i = 0; i < 8; ++i) {
        s0[i] = srow[i + hi8];
        s1[i] = srow[16 + i + hi8];
      }
      v16bf p0 = ldsA(Pb, CK, 0,  0, lane);
      v16bf p1 = ldsA(Pb, CK, 16, 0, lane);
#pragma unroll
      for (int ct = 0; ct < 8; ++ct) {
#pragma unroll
        for (int i = 0; i < 8; ++i) { acc0[ct][i] *= s0[i]; acc1[ct][i] *= s1[i]; }
        v16bf bv = ldsB(VT, CK, c0 + ct * 16, 0, lane);
        acc0[ct] = wmma_bf16(p0, bv, acc0[ct]);
        acc1[ct] = wmma_bf16(p1, bv, acc1[ct]);
      }
    }
    __syncthreads();
  }

  // ---- epilogue: out = residual + acc / l ----
  {
    float li0[8], li1[8];
#pragma unroll
    for (int i = 0; i < 8; ++i) {
      li0[i] = 1.0f / lrow[i + hi8];
      li1[i] = 1.0f / lrow[16 + i + hi8];
    }
#pragma unroll
    for (int ct = 0; ct < 8; ++ct) {
      int col = c0 + ct * 16 + (lane & 15);
#pragma unroll
      for (int i = 0; i < 8; ++i) {
        int r = i + hi8;
        size_t off0 = ((size_t)b * LKEYS + q0 + r) * DD + col;
        Og[off0] = Qg[off0] + acc0[ct][i] * li0[i];
        size_t off1 = ((size_t)b * LKEYS + q0 + 16 + r) * DD + col;
        Og[off1] = Qg[off1] + acc1[ct][i] * li1[i];
      }
    }
  }
}

extern "C" void kernel_launch(void* const* d_in, const int* in_sizes, int n_in,
                              void* d_out, int out_size, void* d_ws, size_t ws_size,
                              hipStream_t stream) {
  const float* S = (const float*)d_in[0];
  const float* D = (const float*)d_in[1];
  float* out = (float*)d_out;
  (void)in_sizes; (void)n_in; (void)out_size; (void)d_ws; (void)ws_size;

  dim3 grid(LKEYS / LQ, 8, 2);   // (q-tiles, batch, direction)
  dim3 block(256);
  size_t shmem =
      (size_t)(LQ * DD + CK * DD + DD * CK) * sizeof(unsigned short)  // Qb, Kb, VT
      + (size_t)2 * LQ * CK * sizeof(float)                           // scp
      + (size_t)LQ * CK * sizeof(unsigned short)                      // Pb
      + (size_t)3 * LQ * sizeof(float);                               // m/l/s rows
  coattn_flash<<<grid, block, shmem, stream>>>(S, D, out);
}
